// MultiheadAttention_22230750724433
// MI455X (gfx1250) — matrix operand order repaired and checked by reading, not compiled, not run on hardware
//
#include <hip/hip_runtime.h>
#include <hip/hip_bf16.h>

typedef __attribute__((ext_vector_type(16))) _Float16 v16h;
typedef __attribute__((ext_vector_type(8)))  _Float16 v8h;
typedef __attribute__((ext_vector_type(8)))  float    v8f;
typedef __attribute__((ext_vector_type(4)))  float    v4f;
typedef __attribute__((ext_vector_type(4)))  int      v4i;

constexpr int MB = 2;      // batch
constexpr int ML = 2048;   // seq len
constexpr int ME = 1024;   // embed
constexpr int MH = 16;     // heads
constexpr int MD = 64;     // head dim

// LDS B-tile row pitch in halves (64B data + 16B pad -> bank spread, 16B aligned)
constexpr int BPITCH = 40;

#if __has_builtin(__builtin_amdgcn_global_load_async_to_lds_b128) && \
    __has_builtin(__builtin_amdgcn_s_wait_asynccnt)
#define MHA_ASYNC 1
#else
#define MHA_ASYNC 0
#endif

__device__ __forceinline__ void stage_b128(const _Float16* gp, _Float16* lp)
{
#if MHA_ASYNC
    // Signature (from hipcc diagnostic): (int4 __device__* src, int4 __shared__* dst,
    // int offset, int cpol). Generic pointers convert implicitly to the CUDA AS
    // qualified params in HIP.
    __builtin_amdgcn_global_load_async_to_lds_b128((v4i*)gp, (v4i*)lp, 0, 0);
#else
    *(v8h*)lp = *(const v8h*)gp;
#endif
}

__device__ __forceinline__ void stage_wait_2()
{
#if MHA_ASYNC
    __builtin_amdgcn_s_wait_asynccnt(2);
#endif
}

__device__ __forceinline__ void stage_wait_0()
{
#if MHA_ASYNC
    __builtin_amdgcn_s_wait_asynccnt(0);
#endif
}

// ---------------------------------------------------------------------------
// fp32 -> f16 elementwise convert
// ---------------------------------------------------------------------------
__global__ __launch_bounds__(256)
void cvt_f32_to_f16(const float* __restrict__ in, _Float16* __restrict__ out, int n)
{
    int i = blockIdx.x * 256 + threadIdx.x;
    if (i < n) out[i] = (_Float16)in[i];
}

// ---------------------------------------------------------------------------
// Generic batched NT GEMM:  C[m,n] = scale * sum_k A[m,k]*B[n,k] (+ bias[n])
// Block: 128 threads = 4 waves; each wave computes a 64x64 tile (4x4 WMMAs).
// Block tile: 256 (M) x 64 (N). The 64-row B tile is shared by all 4 waves,
// so it is staged through LDS with double-buffered async global->LDS copies.
// Batch z decomposed as zb=z/nH, zh=z%nH with independent strides.
// ---------------------------------------------------------------------------
template<bool OUTF16, bool HASBIAS>
__global__ __launch_bounds__(128)
void gemm_nt_wmma(const _Float16* __restrict__ A, long long aSB, long long aSH, int lda,
                  const _Float16* __restrict__ Bm, long long bSB, long long bSH, int ldb,
                  void* __restrict__ Cout, long long cSB, long long cSH, int ldc,
                  const float* __restrict__ bias, float scale, int K, int nH)
{
    __shared__ _Float16 ldsB[2][64 * BPITCH];   // 2 x 5 KB

    const int z  = blockIdx.z;
    const int zb = z / nH;
    const int zh = z % nH;
    const _Float16* __restrict__ Ab = A  + (long long)zb * aSB + (long long)zh * aSH;
    const _Float16* __restrict__ Bb = Bm + (long long)zb * bSB + (long long)zh * bSH;

    const int tid  = threadIdx.x;
    const int wave = tid >> 5;
    const int lane = tid & 31;
    const int g    = lane >> 4;      // half-wave group (K-stripe select)
    const int r    = lane & 15;      // row within tile

    const int m0 = blockIdx.y * 256 + wave * 64;
    const int n0 = blockIdx.x * 64;

    // Cooperative B staging: 64 rows x 32 halves (64B) = 256 b128 segments;
    // 128 threads x 2 segments. Thread t copies row (t>>1), segments
    // {(t&1)*2, (t&1)*2+1}.
    const int srow  = tid >> 1;
    const int sseg0 = (tid & 1) * 2;

    // Prologue: stage k=0 into buffer 0.
    {
        const _Float16* gp = Bb + (long long)(n0 + srow) * ldb + sseg0 * 8;
        _Float16* lp = &ldsB[0][srow * BPITCH + sseg0 * 8];
        stage_b128(gp,     lp);
        stage_b128(gp + 8, lp + 8);
    }

    v8f c[4][4] = {};
    int cur = 0;

    for (int k = 0; k < K; k += 32) {
        const bool hasNext = (k + 32) < K;
        if (hasNext) {
            const _Float16* gp = Bb + (long long)(n0 + srow) * ldb + (k + 32) + sseg0 * 8;
            _Float16* lp = &ldsB[cur ^ 1][srow * BPITCH + sseg0 * 8];
            stage_b128(gp,     lp);
            stage_b128(gp + 8, lp + 8);
            stage_wait_2();          // current buffer's copies (issued earlier) done
        } else {
            stage_wait_0();
        }
        __syncthreads();

        // B fragments from LDS: lane holds row (j*16+r), 16 contiguous halves.
        v16h bfrag[4];
        #pragma unroll
        for (int j = 0; j < 4; ++j) {
            const _Float16* bp = &ldsB[cur][(j * 16 + r) * BPITCH + g * 16];
            bfrag[j] = *(const v16h*)bp;
        }
        #pragma unroll
        for (int i = 0; i < 4; ++i) {
            // A fragment: row (m0+i*16+r), 16 contiguous halves at k+g*16.
            const _Float16* ap = Ab + (long long)(m0 + i * 16 + r) * lda + k + g * 16;
            v8h alo = *(const v8h*)ap;
            v8h ahi = *(const v8h*)(ap + 8);
            if (hasNext) __builtin_prefetch(ap + 32, 0, 1);
            v16h a = __builtin_shufflevector(alo, ahi,
                       0,1,2,3,4,5,6,7,8,9,10,11,12,13,14,15);
            #pragma unroll
            for (int j = 0; j < 4; ++j) {
                c[i][j] = __builtin_amdgcn_wmma_f32_16x16x32_f16(
                    false, a, false, bfrag[j], (short)0, c[i][j], false, false);
            }
        }
        __syncthreads();             // all waves done reading before overwrite
        cur ^= 1;
    }

    const long long cOff = (long long)zb * cSB + (long long)zh * cSH;
    #pragma unroll
    for (int i = 0; i < 4; ++i) {
        #pragma unroll
        for (int j = 0; j < 4; ++j) {
            const int col = n0 + j * 16 + r;
            float bb = 0.0f;
            if constexpr (HASBIAS) bb = bias[col];
            #pragma unroll
            for (int v = 0; v < 8; ++v) {
                const int rowm = m0 + i * 16 + g * 8 + v;
                const float val = c[i][j][v] * scale + bb;
                if constexpr (OUTF16)
                    ((_Float16*)Cout)[cOff + (long long)rowm * ldc + col] = (_Float16)val;
                else
                    ((float*)Cout)[cOff + (long long)rowm * ldc + col] = val;
            }
        }
    }
}

// ---------------------------------------------------------------------------
// Vt[b,h,dd,l] = Vproj[b,l,h*MD+dd]   (f16 -> f16, per-head transpose)
// ---------------------------------------------------------------------------
__global__ __launch_bounds__(256)
void transpose_v(const _Float16* __restrict__ Vp, _Float16* __restrict__ Vt)
{
    const long long total = (long long)MB * MH * MD * ML;
    long long i = (long long)blockIdx.x * 256 + threadIdx.x;
    if (i >= total) return;
    int l  = (int)(i & (ML - 1));
    long long t = i >> 11;            // / ML
    int dd = (int)(t & (MD - 1));
    t >>= 6;                          // / MD
    int h  = (int)(t & (MH - 1));
    int b  = (int)(t >> 4);           // / MH
    Vt[i] = Vp[((long long)(b * ML + l)) * ME + h * MD + dd];
}

// ---------------------------------------------------------------------------
// Row softmax over last dim (ML=2048), in place on fp32 weights.
// One block (256 threads) per row; each thread handles 8 strided elements.
// ---------------------------------------------------------------------------
__global__ __launch_bounds__(256)
void softmax_rows(float* __restrict__ W)
{
    __shared__ float red[256];
    const long long row = blockIdx.x;
    float* __restrict__ p = W + row * (long long)ML;
    const int tid = threadIdx.x;

    float v[8];
    float m = -3.402823466e38f;
    #pragma unroll
    for (int i = 0; i < 8; ++i) {
        v[i] = p[tid + i * 256];
        m = fmaxf(m, v[i]);
    }
    red[tid] = m;
    __syncthreads();
    for (int s = 128; s > 0; s >>= 1) {
        if (tid < s) red[tid] = fmaxf(red[tid], red[tid + s]);
        __syncthreads();
    }
    m = red[0];
    __syncthreads();

    float sum = 0.0f;
    #pragma unroll
    for (int i = 0; i < 8; ++i) {
        v[i] = __expf(v[i] - m);
        sum += v[i];
    }
    red[tid] = sum;
    __syncthreads();
    for (int s = 128; s > 0; s >>= 1) {
        if (tid < s) red[tid] = red[tid] + red[tid + s];
        __syncthreads();
    }
    const float inv = 1.0f / red[0];

    #pragma unroll
    for (int i = 0; i < 8; ++i)
        p[tid + i * 256] = v[i] * inv;
}

// ---------------------------------------------------------------------------
// Z[b, q, h*MD+dd] = sum_k P[b,h,q,k] * Vt[b,h,dd,k]
// P is fp32 (normalized weights in d_out), converted to f16 fragments on the
// fly. NT GEMM per (b,h): M=2048, N=64, K=2048.
// ---------------------------------------------------------------------------
__global__ __launch_bounds__(128)
void attn_pv_wmma(const float* __restrict__ P, long long pSB, long long pSH, int ldp,
                  const _Float16* __restrict__ Vt, long long vSB, long long vSH, int ldv,
                  _Float16* __restrict__ Z, long long zSB, long long zSH, int ldz,
                  int K, int nH)
{
    const int z  = blockIdx.z;
    const int zb = z / nH;
    const int zh = z % nH;
    const float*    __restrict__ Pb = P  + (long long)zb * pSB + (long long)zh * pSH;
    const _Float16* __restrict__ Vb = Vt + (long long)zb * vSB + (long long)zh * vSH;

    const int wave = threadIdx.x >> 5;
    const int lane = threadIdx.x & 31;
    const int g    = lane >> 4;
    const int r    = lane & 15;

    const int m0 = blockIdx.y * 256 + wave * 64;
    const int n0 = 0;  // N = 64 handled by one block column

    v8f c[4][4] = {};

    for (int k = 0; k < K; k += 32) {
        v16h bfrag[4];
        #pragma unroll
        for (int j = 0; j < 4; ++j) {
            const _Float16* bp = Vb + (long long)(n0 + j * 16 + r) * ldv + k + g * 16;
            bfrag[j] = *(const v16h*)bp;
        }
        #pragma unroll
        for (int i = 0; i < 4; ++i) {
            const float* ap = Pb + (long long)(m0 + i * 16 + r) * ldp + k + g * 16;
            v4f f0 = *(const v4f*)ap;
            v4f f1 = *(const v4f*)(ap + 4);
            v4f f2 = *(const v4f*)(ap + 8);
            v4f f3 = *(const v4f*)(ap + 12);
            v16h a;
            #pragma unroll
            for (int t = 0; t < 4; ++t) {
                a[t]      = (_Float16)f0[t];
                a[4 + t]  = (_Float16)f1[t];
                a[8 + t]  = (_Float16)f2[t];
                a[12 + t] = (_Float16)f3[t];
            }
            #pragma unroll
            for (int j = 0; j < 4; ++j) {
                c[i][j] = __builtin_amdgcn_wmma_f32_16x16x32_f16(
                    false, a, false, bfrag[j], (short)0, c[i][j], false, false);
            }
        }
    }

    const long long zOff = (long long)zb * zSB + (long long)zh * zSH;
    #pragma unroll
    for (int i = 0; i < 4; ++i) {
        #pragma unroll
        for (int j = 0; j < 4; ++j) {
            const int col = n0 + j * 16 + r;
            #pragma unroll
            for (int v = 0; v < 8; ++v) {
                const int rowm = m0 + i * 16 + g * 8 + v;
                Z[zOff + (long long)rowm * ldz + col] = (_Float16)c[i][j][v];
            }
        }
    }
}

// ---------------------------------------------------------------------------
// Host-side launch
// ---------------------------------------------------------------------------
extern "C" void kernel_launch(void* const* d_in, const int* in_sizes, int n_in,
                              void* d_out, int out_size, void* d_ws, size_t ws_size,
                              hipStream_t stream)
{
    (void)in_sizes; (void)n_in; (void)out_size; (void)ws_size;

    const float* query = (const float*)d_in[0];
    const float* key   = (const float*)d_in[1];
    const float* value = (const float*)d_in[2];
    const float* Wq    = (const float*)d_in[3];
    const float* bq    = (const float*)d_in[4];
    const float* Wk    = (const float*)d_in[5];
    const float* bk    = (const float*)d_in[6];
    const float* Wv    = (const float*)d_in[7];
    const float* bv    = (const float*)d_in[8];
    const float* Wo    = (const float*)d_in[9];
    const float* bo    = (const float*)d_in[10];

    float* out     = (float*)d_out;                               // [B,L,E]
    float* weights = out + (long long)MB * ML * ME;               // [B,H,L,L]

    // Workspace layout (all 256B aligned by construction)
    char* ws = (char*)d_ws;
    const size_t BLE2 = (size_t)MB * ML * ME * 2;   // 8 MB (f16 [B,L,E])
    const size_t EE2  = (size_t)ME * ME * 2;        // 2 MB (f16 [E,E])
    _Float16* qh  = (_Float16*)(ws);
    _Float16* kh  = (_Float16*)(ws + BLE2);
    _Float16* vh  = (_Float16*)(ws + 2 * BLE2);
    _Float16* wqh = (_Float16*)(ws + 3 * BLE2);
    _Float16* wkh = (_Float16*)(ws + 3 * BLE2 + EE2);
    _Float16* wvh = (_Float16*)(ws + 3 * BLE2 + 2 * EE2);
    _Float16* woh = (_Float16*)(ws + 3 * BLE2 + 3 * EE2);
    _Float16* Qp  = (_Float16*)(ws + 3 * BLE2 + 4 * EE2);
    _Float16* Kp  = (_Float16*)(ws + 4 * BLE2 + 4 * EE2);
    _Float16* Vp  = (_Float16*)(ws + 5 * BLE2 + 4 * EE2);
    _Float16* Vt  = (_Float16*)(ws + 6 * BLE2 + 4 * EE2);
    _Float16* Zh  = (_Float16*)(ws + 7 * BLE2 + 4 * EE2);

    const int nBLE = MB * ML * ME;   // 4194304
    const int nEE  = ME * ME;        // 1048576

    // 1) fp32 -> f16 conversions
    cvt_f32_to_f16<<<(nBLE + 255) / 256, 256, 0, stream>>>(query, qh, nBLE);
    cvt_f32_to_f16<<<(nBLE + 255) / 256, 256, 0, stream>>>(key,   kh, nBLE);
    cvt_f32_to_f16<<<(nBLE + 255) / 256, 256, 0, stream>>>(value, vh, nBLE);
    cvt_f32_to_f16<<<(nEE + 255) / 256, 256, 0, stream>>>(Wq, wqh, nEE);
    cvt_f32_to_f16<<<(nEE + 255) / 256, 256, 0, stream>>>(Wk, wkh, nEE);
    cvt_f32_to_f16<<<(nEE + 255) / 256, 256, 0, stream>>>(Wv, wvh, nEE);
    cvt_f32_to_f16<<<(nEE + 255) / 256, 256, 0, stream>>>(Wo, woh, nEE);

    // 2) Projections: [4096x1024] = [4096x1024] @ W^T + b  (f16 out)
    dim3 gp(ME / 64, (MB * ML) / 256, 1);
    gemm_nt_wmma<true, true><<<gp, 128, 0, stream>>>(
        qh, 0, 0, ME, wqh, 0, 0, ME, (void*)Qp, 0, 0, ME, bq, 1.0f, ME, 1);
    gemm_nt_wmma<true, true><<<gp, 128, 0, stream>>>(
        kh, 0, 0, ME, wkh, 0, 0, ME, (void*)Kp, 0, 0, ME, bk, 1.0f, ME, 1);
    gemm_nt_wmma<true, true><<<gp, 128, 0, stream>>>(
        vh, 0, 0, ME, wvh, 0, 0, ME, (void*)Vp, 0, 0, ME, bv, 1.0f, ME, 1);

    // 3) Per-head V transpose: [B,H,d,L]
    const long long nVt = (long long)MB * MH * MD * ML;  // 4194304
    transpose_v<<<(unsigned)((nVt + 255) / 256), 256, 0, stream>>>(Vp, Vt);

    // 4) Scores: weights[b,h,q,k] = (Q_h @ K_h^T) / sqrt(d)   (fp32 out)
    dim3 gs(ML / 64, ML / 256, MB * MH);
    gemm_nt_wmma<false, false><<<gs, 128, 0, stream>>>(
        Qp, (long long)ML * ME, MD, ME,
        Kp, (long long)ML * ME, MD, ME,
        (void*)weights, (long long)MH * ML * ML, (long long)ML * ML, ML,
        nullptr, 0.125f /* 1/sqrt(64) */, MD, MH);

    // 5) Softmax over keys, in place on weights
    softmax_rows<<<MB * MH * ML, 256, 0, stream>>>(weights);

    // 6) Z = P @ V  (fp32 P from d_out, f16 Vt, f16 Z interleaved back to [B,L,E])
    dim3 gz(1, ML / 256, MB * MH);
    attn_pv_wmma<<<gz, 128, 0, stream>>>(
        weights, (long long)MH * ML * ML, (long long)ML * ML, ML,
        Vt, (long long)MH * MD * ML, (long long)MD * ML, ML,
        Zh, (long long)ML * ME, MD, ME,
        ML, MH);

    // 7) Output projection: out = Z @ Wo^T + bo  (fp32 out)
    gemm_nt_wmma<false, true><<<gp, 128, 0, stream>>>(
        Zh, 0, 0, ME, woh, 0, 0, ME, (void*)out, 0, 0, ME, bo, 1.0f, ME, 1);
}
